// ScannBlock1d_82111184765005
// MI455X (gfx1250) — matrix operand order repaired and checked by reading, not compiled, not run on hardware
//
#include <hip/hip_runtime.h>

typedef __attribute__((ext_vector_type(16))) __bf16 v16bf;
typedef __attribute__((ext_vector_type(8)))  __bf16 v8bf;
typedef __attribute__((ext_vector_type(8)))  float  v8f;

#define B_  4
#define C_  128
#define L_  4096
#define H_  512
#define O_  128
#define CK  32            // keys per flash chunk
#define NCH (L_ / CK)     // 128 chunks

// ---------------- CDNA5 async-copy helpers (gfx1250) ----------------
__device__ __forceinline__ void async_b128(unsigned lds_byte_off, const void* gptr) {
  // GLOBAL_LOAD_ASYNC_TO_LDS_B128, GV mode: vdst = LDS byte addr, vaddr = 64-bit global addr
  asm volatile("global_load_async_to_lds_b128 %0, %1, off"
               :: "v"(lds_byte_off), "v"(gptr) : "memory");
}
__device__ __forceinline__ void wait_async0() {
  asm volatile("s_wait_asynccnt 0" ::: "memory");
}
__device__ __forceinline__ void wait_ds0() {
  asm volatile("s_wait_dscnt 0" ::: "memory");
}
__device__ __forceinline__ unsigned lds_off(const void* p) {
  // low 32 bits of a flat shared-memory address == wave-relative LDS byte offset
  return (unsigned)(unsigned long long)p;
}

__device__ __forceinline__ float halfmax16(float v) {
  #pragma unroll
  for (int m = 1; m <= 8; m <<= 1) v = fmaxf(v, __shfl_xor(v, m, 32));
  return v;
}
__device__ __forceinline__ float halfsum16(float v) {
  #pragma unroll
  for (int m = 1; m <= 8; m <<= 1) v += __shfl_xor(v, m, 32);
  return v;
}

// ---------------- Kernel 1: QKV projections (convs) -> bf16 ----------------
// Q[b,l,h], K[b,l,h]  (row l contiguous in h)   V[b,h,l]  (row h contiguous in l)
__global__ __launch_bounds__(128)
void qkv_kernel(const float* __restrict__ x,
                const float* __restrict__ qw, const float* __restrict__ qb,
                const float* __restrict__ kw, const float* __restrict__ kb,
                const float* __restrict__ vw, const float* __restrict__ vb,
                __bf16* __restrict__ Qm, __bf16* __restrict__ Km, __bf16* __restrict__ Vm)
{
  __shared__ float xs[C_][34];                       // l0-1 .. l0+32 halo
  const int tid = threadIdx.x;
  const int b   = blockIdx.y;
  const int l0  = blockIdx.x * 32;

  for (int e = tid; e < C_ * 34; e += 128) {
    int c = e / 34, j = e % 34;
    int gl = l0 + j - 1;
    xs[c][j] = (gl >= 0 && gl < L_) ? x[((size_t)b * C_ + c) * L_ + gl] : 0.f;
  }
  __syncthreads();

  const int lt = tid & 31, hg = tid >> 5;
  for (int h = hg; h < H_; h += 4) {
    float aq = qb[h], ak = kb[h], av = vb[h];
    const float* wq = qw + (size_t)h * C_ * 3;
    const float* wk = kw + (size_t)h * C_;
    const float* wv = vw + (size_t)h * C_;
    #pragma unroll 4
    for (int c = 0; c < C_; ++c) {
      float xm = xs[c][lt], x0 = xs[c][lt + 1], xp = xs[c][lt + 2];
      aq += xm * wq[3 * c] + x0 * wq[3 * c + 1] + xp * wq[3 * c + 2];
      ak += x0 * wk[c];
      av += x0 * wv[c];
    }
    Qm[((size_t)b * L_ + l0 + lt) * H_ + h] = (__bf16)aq;
    Km[((size_t)b * L_ + l0 + lt) * H_ + h] = (__bf16)ak;
    Vm[((size_t)b * H_ + h) * L_ + l0 + lt] = (__bf16)av;
  }
}

// ---------------- Kernel 2: flash attention with WMMA ----------------
// WG = 4 waves = 64 query rows. Wave w owns rows [w*16, w*16+16) for S/softmax,
// but accumulates O for ALL 64 rows over its own H-quarter (B-reuse x4 in PV).
__global__ __launch_bounds__(128)
void attn_kernel(const __bf16* __restrict__ Qm,   // [B][L][H]
                 const __bf16* __restrict__ Km,   // [B][L][H]
                 const __bf16* __restrict__ Vm,   // [B][H][L]
                 float* __restrict__ Ctx)         // [B][L][H]
{
  __shared__ __bf16 Kt[2][CK][H_];     // 2 x 32KB
  __shared__ __bf16 Vt[2][H_][CK];     // 2 x 32KB
  __shared__ __bf16 Pt[64][CK];        // P for all 64 WG rows, 4KB
  __shared__ float  scl[64];           // per-row scale (and final l) broadcast

  const int tid  = threadIdx.x;
  const int w    = tid >> 5;
  const int lane = tid & 31;
  const int n    = lane & 15;          // column / row index within 16
  const int hi   = lane >> 4;          // lane-half select
  const int b    = blockIdx.x >> 6;    // 64 query tiles per batch
  const int tile = blockIdx.x & 63;
  const int q0   = tile * 64;          // WG base query row

  // ---- Q tile (own 16 rows) resident: 16 A-fragments (16x32 each) ----
  v16bf qa[16];
  {
    const __bf16* qrow = Qm + ((size_t)(b * L_ + q0 + w * 16 + n)) * H_;
    #pragma unroll
    for (int hc = 0; hc < 16; ++hc) {
      union { v16bf v; v8bf h[2]; } u;
      u.h[0] = *(const v8bf*)(qrow + hc * 32 + hi * 16);       // K = 16*hi .. +7
      u.h[1] = *(const v8bf*)(qrow + hc * 32 + hi * 16 + 8);   // K = 16*hi+8 .. +7
      qa[hc] = u.v;
    }
  }

  const v8f vzero = {0.f, 0.f, 0.f, 0.f, 0.f, 0.f, 0.f, 0.f};
  v8f o[8][4];                                // [htile within own H/4][Mtile]
  #pragma unroll
  for (int ht = 0; ht < 8; ++ht)
    #pragma unroll
    for (int mt = 0; mt < 4; ++mt) o[ht][mt] = vzero;
  float m[8], l[8];
  #pragma unroll
  for (int r = 0; r < 8; ++r) { m[r] = -1e30f; l[r] = 0.f; }

  const __bf16* kg0 = Km + (size_t)b * L_ * H_;
  const __bf16* vg0 = Vm + (size_t)b * H_ * L_;

  // ---- prologue: async-stage chunk 0 ----
  #pragma unroll
  for (int i = 0; i < 16; ++i) {
    int idx = i * 128 + tid;                                   // 0..2047 (x16B = 32KB)
    async_b128(lds_off(&Kt[0][0][0]) + idx * 16, (const char*)kg0 + idx * 16);
  }
  #pragma unroll
  for (int i = 0; i < 16; ++i) {
    int idx = i * 128 + tid;
    int h = idx >> 2, seg = idx & 3;
    async_b128(lds_off(&Vt[0][h][seg * 8]), vg0 + (size_t)h * L_ + seg * 8);
  }
  wait_async0();
  __syncthreads();

  #pragma unroll 1
  for (int c = 0; c < NCH; ++c) {
    const int cur = c & 1;
    if (c + 1 < NCH) {                                         // prefetch next chunk
      const int nxt = cur ^ 1;
      const int m1  = (c + 1) * CK;
      const __bf16* kg = kg0 + (size_t)m1 * H_;
      #pragma unroll
      for (int i = 0; i < 16; ++i) {
        int idx = i * 128 + tid;
        async_b128(lds_off(&Kt[nxt][0][0]) + idx * 16, (const char*)kg + idx * 16);
      }
      #pragma unroll
      for (int i = 0; i < 16; ++i) {
        int idx = i * 128 + tid;
        int h = idx >> 2, seg = idx & 3;
        async_b128(lds_off(&Vt[nxt][h][seg * 8]), vg0 + (size_t)h * L_ + m1 + seg * 8);
      }
    }

    // ---- S = Q * K^T for own 16 rows, two 16-key n-tiles, pipelined loads ----
    v8f s0 = vzero, s1 = vzero;
    {
      const __bf16* krow = &Kt[cur][n][hi * 16];               // + hc*32 ; n-tile1 at +16*H_
      v16bf b0 = *(const v16bf*)(krow);
      v16bf b1 = *(const v16bf*)(krow + 16 * H_);
      #pragma unroll
      for (int hc = 0; hc < 16; ++hc) {
        v16bf nb0 = b0, nb1 = b1;
        if (hc < 15) {
          nb0 = *(const v16bf*)(krow + (hc + 1) * 32);
          nb1 = *(const v16bf*)(krow + 16 * H_ + (hc + 1) * 32);
        }
        s0 = __builtin_amdgcn_wmma_f32_16x16x32_bf16(false, qa[hc], false, b0, (short)0, s0, false, false);
        s1 = __builtin_amdgcn_wmma_f32_16x16x32_bf16(false, qa[hc], false, b1, (short)0, s1, false, false);
        b0 = nb0; b1 = nb1;
      }
    }

    // ---- online softmax for own rows (row = w*16 + r + 8*hi) ----
    float scale[8], p0[8], p1[8];
    #pragma unroll
    for (int r = 0; r < 8; ++r) {
      float rm = halfmax16(fmaxf(s0[r], s1[r]));
      float mn = fmaxf(m[r], rm);
      scale[r] = __expf(m[r] - mn);
      p0[r] = __expf(s0[r] - mn);
      p1[r] = __expf(s1[r] - mn);
      float rs = halfsum16(p0[r] + p1[r]);
      l[r] = l[r] * scale[r] + rs;
      m[r] = mn;
    }

    // ---- publish P (bf16, C->A layout staging) and per-row scale ----
    #pragma unroll
    for (int r = 0; r < 8; ++r) {
      Pt[w * 16 + r + 8 * hi][n]      = (__bf16)p0[r];
      Pt[w * 16 + r + 8 * hi][16 + n] = (__bf16)p1[r];
    }
    if (n == 0) {
      #pragma unroll
      for (int r = 0; r < 8; ++r) scl[w * 16 + r + 8 * hi] = scale[r];
    }
    __syncthreads();

    // ---- rescale O (all 64 rows, own H-quarter) ----
    #pragma unroll
    for (int mt = 0; mt < 4; ++mt) {
      float4 sa = *(const float4*)&scl[mt * 16 + 8 * hi];
      float4 sb = *(const float4*)&scl[mt * 16 + 8 * hi + 4];
      float sc8[8] = {sa.x, sa.y, sa.z, sa.w, sb.x, sb.y, sb.z, sb.w};
      #pragma unroll
      for (int ht = 0; ht < 8; ++ht)
        #pragma unroll
        for (int r = 0; r < 8; ++r) o[ht][mt][r] *= sc8[r];
    }

    // ---- load P A-fragments for all 4 M-tiles ----
    v16bf pa[4];
    #pragma unroll
    for (int mt = 0; mt < 4; ++mt) {
      union { v16bf v; v8bf h[2]; } pu;
      pu.h[0] = *(const v8bf*)&Pt[mt * 16 + n][hi * 16];
      pu.h[1] = *(const v8bf*)&Pt[mt * 16 + n][hi * 16 + 8];
      pa[mt] = pu.v;
    }

    // ---- O += P * V : own 8 h-tiles, each V fragment reused by 4 M-tiles ----
    {
      const __bf16* vrow = &Vt[cur][(w * 8) * 16 + n][hi * 16];    // +ht*16*CK per h-tile
      v16bf vb = *(const v16bf*)(vrow);
      #pragma unroll
      for (int ht = 0; ht < 8; ++ht) {
        v16bf nvb = vb;
        if (ht < 7) nvb = *(const v16bf*)(vrow + (ht + 1) * 16 * CK);
        #pragma unroll
        for (int mt = 0; mt < 4; ++mt)
          o[ht][mt] = __builtin_amdgcn_wmma_f32_16x16x32_bf16(false, pa[mt], false, vb, (short)0, o[ht][mt], false, false);
        vb = nvb;
      }
    }

    wait_async0();
    __syncthreads();
  }

  // ---- epilogue: broadcast l, normalize, store ctx [b, l, h] f32 ----
  if (n == 0) {
    #pragma unroll
    for (int r = 0; r < 8; ++r) scl[w * 16 + r + 8 * hi] = l[r];
  }
  __syncthreads();

  float* cb = Ctx + ((size_t)(b * L_ + q0)) * H_;
  #pragma unroll
  for (int mt = 0; mt < 4; ++mt) {
    float4 la = *(const float4*)&scl[mt * 16 + 8 * hi];
    float4 lb = *(const float4*)&scl[mt * 16 + 8 * hi + 4];
    float li[8] = {1.f / la.x, 1.f / la.y, 1.f / la.z, 1.f / la.w,
                   1.f / lb.x, 1.f / lb.y, 1.f / lb.z, 1.f / lb.w};
    #pragma unroll
    for (int ht = 0; ht < 8; ++ht) {
      #pragma unroll
      for (int r = 0; r < 8; ++r) {
        cb[(size_t)(mt * 16 + r + 8 * hi) * H_ + (w * 8 + ht) * 16 + n] = o[ht][mt][r] * li[r];
      }
    }
  }
}

// ---------------- Kernel 3: output 1x1 conv ----------------
__global__ __launch_bounds__(256)
void oproj_kernel(const float* __restrict__ Ctx,   // [B][L][H]
                  const float* __restrict__ ow, const float* __restrict__ ob,
                  float* __restrict__ out)         // [B][O][L]
{
  __shared__ float cs[16][H_];                     // 32KB
  const int tid = threadIdx.x;
  const int b   = blockIdx.y;
  const int l0  = blockIdx.x * 16;

  for (int e = tid; e < 16 * H_; e += 256) {
    int lt = e >> 9, hh = e & 511;
    cs[lt][hh] = Ctx[((size_t)b * L_ + l0 + lt) * H_ + hh];
  }
  __syncthreads();

  for (int idx = tid; idx < O_ * 16; idx += 256) {
    int oc = idx >> 4, lt = idx & 15;
    float acc = ob[oc];
    const float* wr = ow + (size_t)oc * H_;
    #pragma unroll 8
    for (int h = 0; h < H_; ++h) acc += cs[lt][h] * wr[h];
    out[((size_t)b * O_ + oc) * L_ + l0 + lt] = acc;
  }
}

// ---------------- launch ----------------
extern "C" void kernel_launch(void* const* d_in, const int* in_sizes, int n_in,
                              void* d_out, int out_size, void* d_ws, size_t ws_size,
                              hipStream_t stream) {
  const float* x  = (const float*)d_in[0];
  const float* qw = (const float*)d_in[1];
  const float* qb = (const float*)d_in[2];
  const float* kw = (const float*)d_in[3];
  const float* kb = (const float*)d_in[4];
  const float* vw = (const float*)d_in[5];
  const float* vb = (const float*)d_in[6];
  const float* ow = (const float*)d_in[7];
  const float* ob = (const float*)d_in[8];

  char* ws = (char*)d_ws;
  __bf16* Qm = (__bf16*)(ws);                               // 16 MB
  __bf16* Km = (__bf16*)(ws + ((size_t)16 << 20));          // 16 MB
  __bf16* Vm = (__bf16*)(ws + ((size_t)32 << 20));          // 16 MB
  float*  Ctx = (float*)(ws + ((size_t)48 << 20));          // 32 MB

  qkv_kernel<<<dim3(L_ / 32, B_), 128, 0, stream>>>(x, qw, qb, kw, kb, vw, vb, Qm, Km, Vm);
  attn_kernel<<<B_ * (L_ / 64), 128, 0, stream>>>(Qm, Km, Vm, Ctx);
  oproj_kernel<<<dim3(L_ / 16, B_), 256, 0, stream>>>(Ctx, ow, ob, (float*)d_out);
}
